// VQVAEEncoder_67723044323565
// MI455X (gfx1250) — matrix operand order repaired and checked by reading, not compiled, not run on hardware
//
#include <hip/hip_runtime.h>
#include <hip/hip_bf16.h>

// ---------------------------------------------------------------------------
// VQ-VAE encoder for gfx1250 (MI455X): conv1(+ReLU) -> conv2 -> VQ argmin/gather
// GEMM cores use v_wmma_f32_16x16x32_bf16 (bf16 in, f32 accumulate).
// Tile staging uses GLOBAL_LOAD_ASYNC_TO_LDS_B128 (ASYNCcnt) where the copy is
// a pure contiguous move (conv2 A-tiles, VQ A/B tiles, double-buffered in VQ).
// ---------------------------------------------------------------------------

typedef __attribute__((ext_vector_type(16))) __bf16 v16bf;
typedef __attribute__((ext_vector_type(8)))  float  v8f;

__device__ __forceinline__ v8f wmma_bf16(v16bf a, v16bf b, v8f c) {
  // (neg_a, A, neg_b, B, c_mod, C, reuse_a, reuse_b)
  return __builtin_amdgcn_wmma_f32_16x16x32_bf16(false, a, false, b, (short)0, c,
                                                 false, false);
}

__device__ __forceinline__ __bf16 to_bf16(float f) { return (__bf16)f; }

// Async global->LDS 16B copy (per active lane). VDST = LDS byte address; the
// low 32 bits of a generic LDS pointer are the LDS offset (ISA 10.2 aperture
// truncation). Tracked by ASYNCcnt.
__device__ __forceinline__ void async_ld_b128(void* lds, const void* gaddr) {
  unsigned int lds_off = (unsigned int)(unsigned long long)lds;
  asm volatile("global_load_async_to_lds_b128 %0, %1, off"
               :: "v"(lds_off), "v"(gaddr) : "memory");
}
__device__ __forceinline__ void wait_asynccnt0() {
  asm volatile("s_wait_asynccnt 0x0" ::: "memory");
}

// ------------------------------ dims ---------------------------------------
// x: (32, 256, 2048); w1: (1024, 256, 4); w2: (512, 1024, 4); cb: (1024, 512)
#define NB     32
#define C1     256
#define T1     2048
#define HCH    1024
#define T2     1024
#define DCH    512
#define T3     512
#define KCODES 1024

// ---------------------------------------------------------------------------
// f32 -> bf16 convert (for codebook)
// ---------------------------------------------------------------------------
__global__ __launch_bounds__(256) void k_f32_to_bf16(const float* __restrict__ src,
                                                     __bf16* __restrict__ dst, int n) {
  int i = blockIdx.x * 256 + threadIdx.x;
  if (i < n) dst[i] = to_bf16(src[i]);
}

// per-code squared norms (exact, f32)
__global__ __launch_bounds__(256) void k_cnorm(const float* __restrict__ cb,
                                               float* __restrict__ cn) {
  int j = blockIdx.x * 256 + threadIdx.x; // 0..1023
  const float* p = cb + (size_t)j * DCH;
  float s = 0.f;
  for (int d = 0; d < DCH; ++d) { float v = p[d]; s += v * v; }
  cn[j] = s;
}

// ---------------------------------------------------------------------------
// Conv1 + ReLU as implicit GEMM. M = (b,t) positions (32768), N = out ch (1024),
// K = kk-major (kk*256 + c), K=1024. Output h1b layout (b, t, o) bf16 so conv2's
// im2col reads are contiguous in c. (A-tile needs im2col + f32->bf16 convert,
// so loads stay on the VALU path here.)
// ---------------------------------------------------------------------------
__global__ __launch_bounds__(256) void k_conv1(const float* __restrict__ x,
                                               const float* __restrict__ w1,
                                               const float* __restrict__ b1,
                                               __bf16* __restrict__ h1b) {
  __shared__ __bf16 Asm[64][32];   // [m][k]
  __shared__ __bf16 Bsm[128][32];  // [n][k]

  const int tid  = threadIdx.x;
  const int lane = tid & 31;
  const int wave = tid >> 5;
  const int wm   = wave & 3;   // wave m index (16 rows)
  const int wn   = wave >> 2;  // wave n index (64 cols)
  const int hi   = lane >> 4;
  const int ln   = lane & 15;

  const int mbase = blockIdx.x * 64;
  const int nbase = blockIdx.y * 128;

  const int lm    = tid & 63;
  const int lg    = tid >> 6;        // 0..3
  const int pos_n = mbase + lm;
  const int lb    = pos_n >> 10;     // batch
  const int lt    = pos_n & 1023;    // t2
  const int bo = tid & 127;
  const int bg = tid >> 7;           // 0..1

  v8f acc[4] = {};

  for (int kc = 0; kc < 32; ++kc) {
    const int kk  = kc >> 3;          // tap 0..3
    const int c0  = (kc & 7) * 32;    // channel block
    const int pos = 2 * lt + kk - 1;  // stride 2, pad 1
    __syncthreads();
    {  // A tile: im2col of x
      const bool ok = (pos >= 0) && (pos < T1);
      const float* xp = x + (size_t)lb * C1 * T1 + pos;
      #pragma unroll
      for (int i = 0; i < 8; ++i) {
        int c = c0 + lg * 8 + i;
        float v = ok ? xp[(size_t)c * T1] : 0.0f;
        Asm[lm][lg * 8 + i] = to_bf16(v);
      }
    }
    {  // B tile: w1[o, c, kk]
      const float* wp = w1 + (size_t)(nbase + bo) * (C1 * 4) + kk;
      #pragma unroll
      for (int i = 0; i < 16; ++i) {
        int c = c0 + bg * 16 + i;
        Bsm[bo][bg * 16 + i] = to_bf16(wp[c * 4]);
      }
    }
    __syncthreads();

    v16bf afrag;  // 16-bit A 16x32 per-lane layout (ISA 7.12.2)
    {
      const int m = wm * 16 + ln;
      uint4* ap = (uint4*)&afrag;
      ap[0] = *(const uint4*)&Asm[m][hi * 8];
      ap[1] = *(const uint4*)&Asm[m][16 + hi * 8];
    }
    v16bf bfr[4];  // preload all B fragments, then issue WMMAs back-to-back
    #pragma unroll
    for (int tn = 0; tn < 4; ++tn) {
      const int nrow = wn * 64 + tn * 16 + ln;
      uint4* bp = (uint4*)&bfr[tn];
      bp[0] = *(const uint4*)&Bsm[nrow][hi * 8];
      bp[1] = *(const uint4*)&Bsm[nrow][16 + hi * 8];
    }
    #pragma unroll
    for (int tn = 0; tn < 4; ++tn)
      acc[tn] = wmma_bf16(afrag, bfr[tn], acc[tn]);
  }

  // epilogue: +bias, ReLU, store bf16 to h1b[b, t, o]
  #pragma unroll
  for (int tn = 0; tn < 4; ++tn) {
    const int o = nbase + wn * 64 + tn * 16 + ln;
    const float bias = b1[o];
    #pragma unroll
    for (int v = 0; v < 8; ++v) {
      const int mloc = wm * 16 + v + 8 * hi;
      const int n = mbase + mloc;
      const int b = n >> 10, t = n & 1023;
      float val = acc[tn][v] + bias;
      val = val > 0.f ? val : 0.f;
      h1b[(size_t)b * (T2 * HCH) + (size_t)t * HCH + o] = to_bf16(val);
    }
  }
}

// ---------------------------------------------------------------------------
// Conv2 as implicit GEMM. M = (b,t3) (16384), N = out ch (512),
// K = kk*1024 + c (4096). A-tiles are contiguous bf16 rows of h1b -> moved with
// async global->LDS. Writes z_e f32 (b,d,t) and flat bf16 (row=(b,t), col=d).
// ---------------------------------------------------------------------------
__global__ __launch_bounds__(256) void k_conv2(const __bf16* __restrict__ h1b,
                                               const float* __restrict__ w2,
                                               const float* __restrict__ b2,
                                               float* __restrict__ ze_out,
                                               __bf16* __restrict__ flatb) {
  __shared__ __bf16 Asm[64][32];
  __shared__ __bf16 Bsm[128][32];

  const int tid  = threadIdx.x;
  const int lane = tid & 31;
  const int wave = tid >> 5;
  const int wm   = wave & 3;
  const int wn   = wave >> 2;
  const int hi   = lane >> 4;
  const int ln   = lane & 15;

  const int mbase = blockIdx.x * 64;
  const int nbase = blockIdx.y * 128;

  const int lm    = tid & 63;
  const int lg    = tid >> 6;   // 0..3
  const int pos_n = mbase + lm;
  const int lb    = pos_n >> 9;     // batch
  const int lt    = pos_n & 511;    // t3
  const int bo = tid & 127;
  const int bg = tid >> 7;

  v8f acc[4] = {};

  for (int kc = 0; kc < 128; ++kc) {
    const int kk  = kc >> 5;
    const int c0  = (kc & 31) * 32;
    const int pos = 2 * lt + kk - 1;
    __syncthreads();
    {  // A tile: h1b[b, pos, c0..] contiguous 16B -> async copy to LDS
      if (pos >= 0 && pos < T2) {
        async_ld_b128(&Asm[lm][lg * 8],
                      h1b + (size_t)lb * (T2 * HCH) + (size_t)pos * HCH + c0 + lg * 8);
      } else {
        *(uint4*)&Asm[lm][lg * 8] = make_uint4(0u, 0u, 0u, 0u);
      }
    }
    {  // B tile: w2[o, c, kk] (strided f32 -> bf16 convert, VALU path)
      const float* wp = w2 + (size_t)(nbase + bo) * (HCH * 4) + kk;
      #pragma unroll
      for (int i = 0; i < 16; ++i) {
        int c = c0 + bg * 16 + i;
        Bsm[bo][bg * 16 + i] = to_bf16(wp[c * 4]);
      }
    }
    wait_asynccnt0();
    __syncthreads();

    v16bf afrag;
    {
      const int m = wm * 16 + ln;
      uint4* ap = (uint4*)&afrag;
      ap[0] = *(const uint4*)&Asm[m][hi * 8];
      ap[1] = *(const uint4*)&Asm[m][16 + hi * 8];
    }
    v16bf bfr[4];
    #pragma unroll
    for (int tn = 0; tn < 4; ++tn) {
      const int nrow = wn * 64 + tn * 16 + ln;
      uint4* bp = (uint4*)&bfr[tn];
      bp[0] = *(const uint4*)&Bsm[nrow][hi * 8];
      bp[1] = *(const uint4*)&Bsm[nrow][16 + hi * 8];
    }
    #pragma unroll
    for (int tn = 0; tn < 4; ++tn)
      acc[tn] = wmma_bf16(afrag, bfr[tn], acc[tn]);
  }

  #pragma unroll
  for (int tn = 0; tn < 4; ++tn) {
    const int o = nbase + wn * 64 + tn * 16 + ln;
    const float bias = b2[o];
    #pragma unroll
    for (int v = 0; v < 8; ++v) {
      const int mloc = wm * 16 + v + 8 * hi;
      const int n = mbase + mloc;            // row = b*512 + t
      const int b = n >> 9, t = n & 511;
      float val = acc[tn][v] + bias;
      ze_out[(size_t)b * (DCH * T3) + (size_t)o * T3 + t] = val;  // z_e (b,d,t)
      flatb[(size_t)n * DCH + o] = to_bf16(val);                  // flat (r,d)
    }
  }
}

// ---------------------------------------------------------------------------
// VQ: per 64 flat rows, loop over 8 chunks of 128 codes; dot = flat @ cb^T via
// WMMA; score = |e|^2 - 2*dot (|z|^2 constant per row -> dropped from argmin).
// Tiles are pure contiguous copies -> double-buffered async global->LDS so the
// k+1 copy overlaps the k WMMAs (one barrier per k-step). (score,idx) reduced
// with monotone-key u64 atomicMin in LDS (ds_min_u64), then gather z_q rows
// from the f32 codebook (raw reshape, per reference).
// ---------------------------------------------------------------------------
__global__ __launch_bounds__(256) void k_vq(const __bf16* __restrict__ flatb,
                                            const __bf16* __restrict__ cbb,
                                            const float* __restrict__ cnorm,
                                            const float* __restrict__ codebook,
                                            float* __restrict__ zq_out,
                                            float* __restrict__ idx_out) {
  __shared__ __bf16 Asm[2][64][32];
  __shared__ __bf16 Bsm[2][128][32];
  __shared__ unsigned long long best[64];

  const int tid  = threadIdx.x;
  const int lane = tid & 31;
  const int wave = tid >> 5;
  const int wm   = wave & 3;
  const int wn   = wave >> 2;
  const int hi   = lane >> 4;
  const int ln   = lane & 15;

  const int mbase = blockIdx.x * 64;

  if (tid < 64) best[tid] = 0xFFFFFFFFFFFFFFFFull;

  const int lm = tid & 63;
  const int lg = tid >> 6;
  const int bo = tid & 127;
  const int bg = tid >> 7;

  const __bf16* arow = flatb + (size_t)(mbase + lm) * DCH + lg * 8;

  for (int nb = 0; nb < 8; ++nb) {        // code chunks of 128
    const __bf16* brow = cbb + (size_t)(nb * 128 + bo) * DCH + bg * 16;
    v8f acc[4] = {};
    __syncthreads();  // prior consumers of buffer 0 are done
    // prologue: stage k-step 0 into buffer 0
    async_ld_b128(&Asm[0][lm][lg * 8], arow);
    async_ld_b128(&Bsm[0][bo][bg * 16], brow);
    async_ld_b128(&Bsm[0][bo][bg * 16 + 8], brow + 8);

    for (int kc = 0; kc < 16; ++kc) {     // K = 512 in steps of 32
      const int cur = kc & 1;
      wait_asynccnt0();
      __syncthreads();                    // tiles for `cur` landed in all waves
      if (kc + 1 < 16) {                  // stage k-step kc+1 into other buffer
        const int d1 = (kc + 1) * 32;
        async_ld_b128(&Asm[cur ^ 1][lm][lg * 8], arow + d1);
        async_ld_b128(&Bsm[cur ^ 1][bo][bg * 16], brow + d1);
        async_ld_b128(&Bsm[cur ^ 1][bo][bg * 16 + 8], brow + d1 + 8);
      }

      v16bf afrag;
      {
        const int m = wm * 16 + ln;
        uint4* ap = (uint4*)&afrag;
        ap[0] = *(const uint4*)&Asm[cur][m][hi * 8];
        ap[1] = *(const uint4*)&Asm[cur][m][16 + hi * 8];
      }
      v16bf bfr[4];
      #pragma unroll
      for (int tn = 0; tn < 4; ++tn) {
        const int nrow = wn * 64 + tn * 16 + ln;
        uint4* bp = (uint4*)&bfr[tn];
        bp[0] = *(const uint4*)&Bsm[cur][nrow][hi * 8];
        bp[1] = *(const uint4*)&Bsm[cur][nrow][16 + hi * 8];
      }
      #pragma unroll
      for (int tn = 0; tn < 4; ++tn)
        acc[tn] = wmma_bf16(afrag, bfr[tn], acc[tn]);
    }
    // chunk epilogue: in-lane min over 4 tiles, then LDS u64 atomicMin per row
    #pragma unroll
    for (int v = 0; v < 8; ++v) {
      float bestv = 3.4e38f;
      int   bestj = 0;
      #pragma unroll
      for (int tn = 0; tn < 4; ++tn) {
        const int j = nb * 128 + wn * 64 + tn * 16 + ln;
        const float s = cnorm[j] - 2.0f * acc[tn][v];
        if (s < bestv) { bestv = s; bestj = j; }
      }
      unsigned int u   = __float_as_uint(bestv);
      unsigned int key = (u & 0x80000000u) ? ~u : (u | 0x80000000u);
      unsigned long long pack =
          ((unsigned long long)key << 32) | (unsigned int)bestj;
      const int row = wm * 16 + v + 8 * hi;
      atomicMin(&best[row], pack);
    }
  }
  __syncthreads();

  if (tid < 64)
    idx_out[mbase + tid] = (float)(unsigned int)(best[tid] & 0xFFFFFFFFu);

  // gather: z_q flat[r*512 + d] = codebook[idx_r][d]  (raw reshape semantics)
  for (int e = tid; e < 64 * DCH; e += 256) {
    const int row = e >> 9, d = e & 511;
    const unsigned int j = (unsigned int)(best[row] & 0xFFFFFFFFu);
    zq_out[(size_t)(mbase + row) * DCH + d] = codebook[(size_t)j * DCH + d];
  }
}

// ---------------------------------------------------------------------------
extern "C" void kernel_launch(void* const* d_in, const int* in_sizes, int n_in,
                              void* d_out, int out_size, void* d_ws, size_t ws_size,
                              hipStream_t stream) {
  (void)in_sizes; (void)n_in; (void)out_size; (void)ws_size;
  const float* x  = (const float*)d_in[0];
  const float* w1 = (const float*)d_in[1];
  const float* b1 = (const float*)d_in[2];
  const float* w2 = (const float*)d_in[3];
  const float* b2 = (const float*)d_in[4];
  const float* cb = (const float*)d_in[5];

  float* out  = (float*)d_out;
  float* zq   = out;                              // 32*512*512
  float* idxf = out + (size_t)NB * DCH * T3;      // 16384 (indices as float)
  float* ze   = idxf + (size_t)NB * T3;           // 32*512*512

  // workspace layout (~81 MB)
  char* ws = (char*)d_ws;
  __bf16* cbb   = (__bf16*)(ws);                       // 1024*512 bf16 = 1 MB
  float*  cn    = (float*)(ws + (1u << 20));           // 1024 f32
  __bf16* h1b   = (__bf16*)(ws + (1u << 20) + 4096);   // 32*1024*1024 bf16 = 64 MB
  __bf16* flatb = (__bf16*)(ws + (1u << 20) + 4096 + ((size_t)NB * T2 * HCH * 2)); // 16 MB

  // codebook -> bf16, code norms
  k_f32_to_bf16<<<(KCODES * DCH + 255) / 256, 256, 0, stream>>>(cb, cbb, KCODES * DCH);
  k_cnorm<<<KCODES / 256, 256, 0, stream>>>(cb, cn);

  // conv1 + ReLU: M=32768 (blocks of 64), N=1024 (blocks of 128)
  k_conv1<<<dim3(512, 8, 1), 256, 0, stream>>>(x, w1, b1, h1b);

  // conv2: M=16384, N=512
  k_conv2<<<dim3(256, 4, 1), 256, 0, stream>>>(h1b, w2, b2, ze, flatb);

  // VQ argmin + gather: 256 blocks of 64 rows
  k_vq<<<dim3(256, 1, 1), 256, 0, stream>>>(flatb, cbb, cn, cb, zq, idxf);
}
